// StandardVQ_13975823581590
// MI455X (gfx1250) — compile-verified, matrix operand order checked by reading, not run on hardware
//
#include <hip/hip_runtime.h>
#include <cstdint>
#include <cstddef>

// ---------------------------------------------------------------------------
// VQ-VAE forward for MI455X (gfx1250): bf16 WMMA GEMM pipeline, v2.1.
//  - All B operands pre-transposed to [N][K] bf16 so both A and B tiles are
//    loaded with contiguous global_load_b128 (no per-element address mults).
//  - GEMM templated on compile-time N, K -> immediate-folded addressing.
// ---------------------------------------------------------------------------

typedef __attribute__((ext_vector_type(16))) __bf16 v16bf;
typedef __attribute__((ext_vector_type(8)))  float  v8f;

static constexpr int BATCH = 8192;
static constexpr int DIN   = 1024;
static constexpr int HID   = 2048;
static constexpr int LAT   = 256;
static constexpr int KCB   = 4096;

struct Q2 { uint4 a; uint4 b; };   // 32 bytes == v16bf

__device__ __forceinline__ unsigned short f2bf(float f) {
  unsigned u = __float_as_uint(f);
  u = u + 0x7FFFu + ((u >> 16) & 1u);      // round-to-nearest-even
  return (unsigned short)(u >> 16);
}

__device__ __forceinline__ float gelu_exact(float x) {
  return 0.5f * x * (1.0f + erff(x * 0.70710678118654752f));
}

// order-preserving float -> uint encoding (min of encoding == min of float)
__device__ __forceinline__ unsigned fenc(float f) {
  unsigned u = __float_as_uint(f);
  return (u & 0x80000000u) ? ~u : (u | 0x80000000u);
}

// ---------------------------------------------------------------------------
// Elementwise f32 -> bf16 convert
// ---------------------------------------------------------------------------
__global__ void k_convert_bf16(const float* __restrict__ s,
                               unsigned short* __restrict__ d, int n) {
  int i = blockIdx.x * blockDim.x + threadIdx.x;
  int stride = gridDim.x * blockDim.x;
  for (; i < n; i += stride) d[i] = f2bf(s[i]);
}

// Transpose + convert: S [R][C] f32 -> T [C][R] bf16, LDS-tiled, coalesced.
__global__ void k_cvt_transpose(const float* __restrict__ S,
                                unsigned short* __restrict__ T, int R, int C) {
  __shared__ float tile[32][33];
  int c0 = blockIdx.x * 32, r0 = blockIdx.y * 32;
  int tx = threadIdx.x, ty = threadIdx.y;   // block (32, 8)
#pragma unroll
  for (int i = ty; i < 32; i += 8)
    tile[i][tx] = S[(size_t)(r0 + i) * C + c0 + tx];
  __syncthreads();
#pragma unroll
  for (int i = ty; i < 32; i += 8)
    T[(size_t)(c0 + i) * R + r0 + tx] = f2bf(tile[tx][i]);
}

// esq[k] = ||embed[k]||^2  (f32, exact)
__global__ void k_rowsq(const float* __restrict__ e, float* __restrict__ esq) {
  __shared__ float sm[256];
  int r = blockIdx.x;
  float v = e[(size_t)r * LAT + threadIdx.x];
  sm[threadIdx.x] = v * v;
  __syncthreads();
  for (int s = 128; s > 0; s >>= 1) {
    if (threadIdx.x < s) sm[threadIdx.x] += sm[threadIdx.x + s];
    __syncthreads();
  }
  if (threadIdx.x == 0) esq[r] = sm[0];
}

__global__ void k_init_rowmin(unsigned long long* __restrict__ rm) {
  int i = blockIdx.x * blockDim.x + threadIdx.x;
  if (i < BATCH) rm[i] = ~0ull;
}

// ---------------------------------------------------------------------------
// Tiled bf16 WMMA GEMM: C[M,N] = A[M,K] @ BT[N,K]^T   (A, BT bf16 row-major)
//   block = 256 threads (8 wave32), tile 128x128, wave tile 32x64,
//   K-step 32 with double-buffered LDS, global_prefetch 2 steps ahead.
//   Epilogue by MODE:
//   0: outb = bf16(gelu(v + aux[n]))
//   1: outf = v + aux[n]; outb = bf16(same)                   (z_e)
//   2: score = aux[n] - 2v; row argmin -> atomicMin(rowmin)   (codebook)
//   3: outf = v + aux[n]                                      (x_recon)
// ---------------------------------------------------------------------------
template <int MODE, int N, int Kd>
__global__ __launch_bounds__(256, 2) void k_gemm(
    const unsigned short* __restrict__ A, const unsigned short* __restrict__ BT,
    const float* __restrict__ aux,
    unsigned short* __restrict__ outb, float* __restrict__ outf,
    unsigned long long* __restrict__ rowmin) {
  // Both tiles 128(rows/cols) x 32(k), stride 40 halves -> 16B-aligned rows.
  __shared__ unsigned short As[2][128 * 40];
  __shared__ unsigned short Bs[2][128 * 40];

  const int tid  = threadIdx.x;
  const int lane = tid & 31;
  const int wave = tid >> 5;
  const int l15  = lane & 15;
  const int hi   = lane >> 4;          // 0: K 0-7/16-23, 1: K 8-15/24-31
  const int wm   = (wave & 3) * 32;    // wave row offset in tile
  const int wn   = (wave >> 2) * 64;   // wave col offset in tile
  const int m0   = blockIdx.y * 128;
  const int n0   = blockIdx.x * 128;

  // Per-thread tile-chunk coordinates (512 16B chunks, 2 per thread).
  const int rowc0 = tid >> 2, seg0 = (tid & 3) * 8;
  const int rowc1 = (tid + 256) >> 2, seg1 = ((tid + 256) & 3) * 8;
  const unsigned short* Ab0 = A + (size_t)(m0 + rowc0) * Kd + seg0;
  const unsigned short* Ab1 = A + (size_t)(m0 + rowc1) * Kd + seg1;
  const unsigned short* Bb0 = BT + (size_t)(n0 + rowc0) * Kd + seg0;
  const unsigned short* Bb1 = BT + (size_t)(n0 + rowc1) * Kd + seg1;

  v8f zero = {};
  v8f acc[2][4];
#pragma unroll
  for (int i = 0; i < 2; ++i)
#pragma unroll
    for (int j = 0; j < 4; ++j) acc[i][j] = zero;

  auto loadTiles = [&](int buf, int k0) {
    uint4 a0 = *(const uint4*)(Ab0 + k0);
    uint4 a1 = *(const uint4*)(Ab1 + k0);
    uint4 b0 = *(const uint4*)(Bb0 + k0);
    uint4 b1 = *(const uint4*)(Bb1 + k0);
    *(uint4*)(&As[buf][rowc0 * 40 + seg0]) = a0;
    *(uint4*)(&As[buf][rowc1 * 40 + seg1]) = a1;
    *(uint4*)(&Bs[buf][rowc0 * 40 + seg0]) = b0;
    *(uint4*)(&Bs[buf][rowc1 * 40 + seg1]) = b1;
  };

  constexpr int nK = Kd >> 5;
  loadTiles(0, 0);
  if (nK > 2) {   // warm L2/L0 for the step after next
    __builtin_prefetch(Ab0 + 64, 0, 1);
    __builtin_prefetch(Bb0 + 64, 0, 1);
  }
  __syncthreads();

  for (int kt = 0; kt < nK; ++kt) {
    const int cur = kt & 1;
    if (kt + 1 < nK) loadTiles(cur ^ 1, (kt + 1) << 5);
    if (kt + 2 < nK) {
      __builtin_prefetch(Ab0 + ((kt + 2) << 5), 0, 1);
      __builtin_prefetch(Bb0 + ((kt + 2) << 5), 0, 1);
    }
    const unsigned short* Ac = As[cur];
    const unsigned short* Bc = Bs[cur];
    const int kb = hi * 8;

    v16bf af[2], bfr[4];
#pragma unroll
    for (int mt = 0; mt < 2; ++mt) {
      int r = wm + mt * 16 + l15;
      Q2 q;
      q.a = *(const uint4*)(Ac + r * 40 + kb);       // K kb..kb+7
      q.b = *(const uint4*)(Ac + r * 40 + 16 + kb);  // K 16+kb..16+kb+7
      af[mt] = __builtin_bit_cast(v16bf, q);
    }
#pragma unroll
    for (int nt = 0; nt < 4; ++nt) {
      int c = wn + nt * 16 + l15;
      Q2 q;
      q.a = *(const uint4*)(Bc + c * 40 + kb);
      q.b = *(const uint4*)(Bc + c * 40 + 16 + kb);
      bfr[nt] = __builtin_bit_cast(v16bf, q);
    }
#pragma unroll
    for (int mt = 0; mt < 2; ++mt)
#pragma unroll
      for (int nt = 0; nt < 4; ++nt)
        acc[mt][nt] = __builtin_amdgcn_wmma_f32_16x16x32_bf16(
            false, af[mt], false, bfr[nt], (short)0, acc[mt][nt], false, false);
    __syncthreads();
  }

  if (MODE == 2) {
    // fused codebook-distance argmin epilogue
#pragma unroll
    for (int mt = 0; mt < 2; ++mt) {
      float es[4];
      int cols[4];
#pragma unroll
      for (int nt = 0; nt < 4; ++nt) {
        cols[nt] = n0 + wn + nt * 16 + l15;
        es[nt] = aux[cols[nt]];
      }
#pragma unroll
      for (int r = 0; r < 8; ++r) {
        unsigned long long best = ~0ull;
#pragma unroll
        for (int nt = 0; nt < 4; ++nt) {
          float score = es[nt] - 2.0f * acc[mt][nt][r];
          unsigned long long e =
              ((unsigned long long)fenc(score) << 32) | (unsigned)cols[nt];
          best = (e < best) ? e : best;
        }
        // lanes 0-15 (and 16-31) each hold one column-candidate of SAME row
#pragma unroll
        for (int off = 1; off < 16; off <<= 1) {
          unsigned long long o = __shfl_xor(best, off, 16);
          best = (o < best) ? o : best;
        }
        if (l15 == 0) {
          int gm = m0 + wm + mt * 16 + r + 8 * hi;
          atomicMin(rowmin + gm, best);
        }
      }
    }
  } else {
#pragma unroll
    for (int mt = 0; mt < 2; ++mt)
#pragma unroll
      for (int nt = 0; nt < 4; ++nt) {
        int gn = n0 + wn + nt * 16 + l15;
        float bv = aux[gn];
#pragma unroll
        for (int r = 0; r < 8; ++r) {
          int gm = m0 + wm + mt * 16 + r + 8 * hi;
          float v = acc[mt][nt][r] + bv;
          if (MODE == 0) {
            outb[(size_t)gm * N + gn] = f2bf(gelu_exact(v));
          } else if (MODE == 1) {
            outf[(size_t)gm * N + gn] = v;
            outb[(size_t)gm * N + gn] = f2bf(v);
          } else {
            outf[(size_t)gm * N + gn] = v;
          }
        }
      }
  }
}

// ---------------------------------------------------------------------------
__global__ void k_decode(const unsigned long long* __restrict__ rm,
                         int* __restrict__ idx, float* __restrict__ oidx) {
  int i = blockIdx.x * blockDim.x + threadIdx.x;
  if (i < BATCH) {
    int id = (int)(unsigned)(rm[i] & 0xFFFFFFFFull);
    idx[i] = id;
    oidx[i] = (float)id;   // indices slot of the output tuple
  }
}

// z_q gather (bf16 for decoder) + per-row squared-error partial
__global__ void k_gather_loss(const float* __restrict__ embed,
                              const int* __restrict__ idx,
                              const float* __restrict__ zef,
                              unsigned short* __restrict__ zqb,
                              float* __restrict__ part) {
  __shared__ float sm[256];
  int b = blockIdx.x, l = threadIdx.x;
  int id = idx[b];
  float e = embed[(size_t)id * LAT + l];
  float z = zef[(size_t)b * LAT + l];
  float d = z - e;
  zqb[(size_t)b * LAT + l] = f2bf(e);
  sm[l] = d * d;
  __syncthreads();
  for (int s = 128; s > 0; s >>= 1) {
    if (l < s) sm[l] += sm[l + s];
    __syncthreads();
  }
  if (l == 0) part[b] = sm[0];
}

// vq_loss = codebook + 0.25*commitment == 1.25 * mean((z_e - z_q)^2) forward
__global__ void k_loss(const float* __restrict__ part, float* __restrict__ out) {
  __shared__ float sm[256];
  float s = 0.0f;
  for (int i = threadIdx.x; i < BATCH; i += 256) s += part[i];
  sm[threadIdx.x] = s;
  __syncthreads();
  for (int t = 128; t > 0; t >>= 1) {
    if (threadIdx.x < t) sm[threadIdx.x] += sm[threadIdx.x + t];
    __syncthreads();
  }
  if (threadIdx.x == 0)
    out[0] = 1.25f * sm[0] / (float)((size_t)BATCH * LAT);
}

// ---------------------------------------------------------------------------
extern "C" void kernel_launch(void* const* d_in, const int* in_sizes, int n_in,
                              void* d_out, int out_size, void* d_ws,
                              size_t ws_size, hipStream_t stream) {
  const float* x  = (const float*)d_in[0];
  const float* W1 = (const float*)d_in[1];
  const float* b1 = (const float*)d_in[2];
  const float* W2 = (const float*)d_in[3];
  const float* b2 = (const float*)d_in[4];
  const float* W3 = (const float*)d_in[5];
  const float* b3 = (const float*)d_in[6];
  const float* EM = (const float*)d_in[7];
  const float* D1 = (const float*)d_in[8];
  const float* e1 = (const float*)d_in[9];
  const float* D2 = (const float*)d_in[10];
  const float* e2 = (const float*)d_in[11];
  const float* D3 = (const float*)d_in[12];
  const float* e3 = (const float*)d_in[13];
  float* out = (float*)d_out;

  char* ws = (char*)d_ws;
  size_t off = 0;
  auto alloc = [&](size_t bytes) -> void* {
    void* p = ws + off;
    off += (bytes + 255) & ~(size_t)255;
    return p;
  };
  unsigned short* xb   = (unsigned short*)alloc((size_t)BATCH * DIN * 2);
  unsigned short* w1t  = (unsigned short*)alloc((size_t)DIN * HID * 2);
  unsigned short* w2t  = (unsigned short*)alloc((size_t)HID * HID * 2);
  unsigned short* w3t  = (unsigned short*)alloc((size_t)HID * LAT * 2);
  unsigned short* emb  = (unsigned short*)alloc((size_t)KCB * LAT * 2);
  unsigned short* d1t  = (unsigned short*)alloc((size_t)LAT * HID * 2);
  unsigned short* d2t  = (unsigned short*)alloc((size_t)HID * HID * 2);
  unsigned short* d3t  = (unsigned short*)alloc((size_t)HID * DIN * 2);
  unsigned short* acta = (unsigned short*)alloc((size_t)BATCH * HID * 2);
  unsigned short* actb = (unsigned short*)alloc((size_t)BATCH * HID * 2);
  float*          zef  = (float*)alloc((size_t)BATCH * LAT * 4);
  unsigned short* zeb  = (unsigned short*)alloc((size_t)BATCH * LAT * 2);
  unsigned short* zqb  = (unsigned short*)alloc((size_t)BATCH * LAT * 2);
  float*          esq  = (float*)alloc((size_t)KCB * 4);
  unsigned long long* rowmin = (unsigned long long*)alloc((size_t)BATCH * 8);
  int*            idx  = (int*)alloc((size_t)BATCH * 4);
  float*          part = (float*)alloc((size_t)BATCH * 4);

  // A operands: plain bf16 convert
  k_convert_bf16<<<(BATCH * DIN + 255) / 256, 256, 0, stream>>>(x, xb,
                                                               BATCH * DIN);
  // dist-GEMM B operand BT[n][k] == embed[n][k]: plain convert, no transpose
  k_convert_bf16<<<(KCB * LAT + 255) / 256, 256, 0, stream>>>(EM, emb,
                                                              KCB * LAT);
  // weight B operands: transpose-convert W[K][N] -> WT[N][K]
  dim3 tb(32, 8);
  k_cvt_transpose<<<dim3(HID / 32, DIN / 32), tb, 0, stream>>>(W1, w1t, DIN, HID);
  k_cvt_transpose<<<dim3(HID / 32, HID / 32), tb, 0, stream>>>(W2, w2t, HID, HID);
  k_cvt_transpose<<<dim3(LAT / 32, HID / 32), tb, 0, stream>>>(W3, w3t, HID, LAT);
  k_cvt_transpose<<<dim3(HID / 32, LAT / 32), tb, 0, stream>>>(D1, d1t, LAT, HID);
  k_cvt_transpose<<<dim3(HID / 32, HID / 32), tb, 0, stream>>>(D2, d2t, HID, HID);
  k_cvt_transpose<<<dim3(DIN / 32, HID / 32), tb, 0, stream>>>(D3, d3t, HID, DIN);

  k_rowsq<<<KCB, 256, 0, stream>>>(EM, esq);
  k_init_rowmin<<<BATCH / 256, 256, 0, stream>>>(rowmin);

  dim3 blk(256);
  // encoder
  k_gemm<0, HID, DIN><<<dim3(HID / 128, BATCH / 128), blk, 0, stream>>>(
      xb, w1t, b1, acta, nullptr, nullptr);
  k_gemm<0, HID, HID><<<dim3(HID / 128, BATCH / 128), blk, 0, stream>>>(
      acta, w2t, b2, actb, nullptr, nullptr);
  k_gemm<1, LAT, HID><<<dim3(LAT / 128, BATCH / 128), blk, 0, stream>>>(
      actb, w3t, b3, zeb, zef, nullptr);
  // codebook distance + fused argmin
  k_gemm<2, KCB, LAT><<<dim3(KCB / 128, BATCH / 128), blk, 0, stream>>>(
      zeb, emb, esq, nullptr, nullptr, rowmin);
  k_decode<<<BATCH / 256, 256, 0, stream>>>(rowmin, idx,
                                            out + (size_t)BATCH * DIN + 1);
  k_gather_loss<<<BATCH, 256, 0, stream>>>(EM, idx, zef, zqb, part);
  k_loss<<<1, 256, 0, stream>>>(part, out + (size_t)BATCH * DIN);
  // decoder (z_st forward value == z_q)
  k_gemm<0, HID, LAT><<<dim3(HID / 128, BATCH / 128), blk, 0, stream>>>(
      zqb, d1t, e1, acta, nullptr, nullptr);
  k_gemm<0, HID, HID><<<dim3(HID / 128, BATCH / 128), blk, 0, stream>>>(
      acta, d2t, e2, actb, nullptr, nullptr);
  k_gemm<3, DIN, HID><<<dim3(DIN / 128, BATCH / 128), blk, 0, stream>>>(
      actb, d3t, e3, nullptr, out, nullptr);
}
